// MyRNN_86208583565846
// MI455X (gfx1250) — matrix operand order repaired and checked by reading, not compiled, not run on hardware
//
#include <hip/hip_runtime.h>
#include <hip/hip_bf16.h>

#define T_STEPS 2048
#define MB      64
#define IN_DIM  256
#define H_DIM   256

typedef __bf16 bf16x8  __attribute__((ext_vector_type(8)));
typedef __bf16 bf16x16 __attribute__((ext_vector_type(16)));
typedef float  f32x4   __attribute__((ext_vector_type(4)));
typedef float  f32x8   __attribute__((ext_vector_type(8)));

#if defined(__has_builtin)
#if __has_builtin(__builtin_amdgcn_tanhf)
#define HAVE_HW_TANH 1
#endif
#endif

__device__ __forceinline__ float fast_tanh(float x) {
#ifdef HAVE_HW_TANH
    return __builtin_amdgcn_tanhf(x);   // v_tanh_f32 (gfx1250 trans op)
#else
    return tanhf(x);
#endif
}

__device__ __forceinline__ bf16x16 cat8(bf16x8 lo, bf16x8 hi) {
    return __builtin_shufflevector(lo, hi, 0,1,2,3,4,5,6,7,8,9,10,11,12,13,14,15);
}

// ---------------------------------------------------------------------------
// Kernel 0: convert weights f32 -> bf16 into workspace.
// ws layout: [0, H*IN)  = Wih (bf16),  [H*IN, 2*H*IN) = Whh (bf16)
// ---------------------------------------------------------------------------
__global__ void cvt_weights_kernel(const float* __restrict__ wih,
                                   const float* __restrict__ whh,
                                   unsigned short* __restrict__ dst) {
    int i = blockIdx.x * blockDim.x + threadIdx.x;
    if (i < H_DIM * IN_DIM) {
        dst[i] = __builtin_bit_cast(unsigned short, (__bf16)wih[i]);
    } else if (i < 2 * H_DIM * IN_DIM) {
        dst[i] = __builtin_bit_cast(unsigned short, (__bf16)whh[i - H_DIM * IN_DIM]);
    }
}

// ---------------------------------------------------------------------------
// Kernel 1 (parallel over t): a_t = x_t @ Wih^T + (Wih_b + Whh_b) -> out[t]
// 512 threads = 16 waves; wave owns 2x2 grid of 16x16 tiles of the 64x256 slab.
// ---------------------------------------------------------------------------
__global__ __launch_bounds__(512, 4) void rnn_phase1_kernel(
        const float* __restrict__ x,
        const unsigned short* __restrict__ wih_u16,
        const float* __restrict__ bih,
        const float* __restrict__ bhh,
        float* __restrict__ out) {
    const __bf16* __restrict__ wih_bf = reinterpret_cast<const __bf16*>(wih_u16);

    const int t    = blockIdx.x;
    const int wave = threadIdx.x >> 5;
    const int lane = threadIdx.x & 31;
    const int lh   = lane & 15;
    const int half = lane >> 4;
    const int m0   = (wave >> 3) * 2;   // tile-row base: 0 or 2 (MB = 4 tiles)
    const int n0   = (wave & 7) * 2;    // tile-col base: 0..14 (H = 16 tiles)

    // Preload B fragments (B[k,n] = Wih[n,k]: table order, 2x8 bf16 per lane).
    bf16x16 Bf[2][8];
#pragma unroll
    for (int nt = 0; nt < 2; ++nt) {
        const int col = (n0 + nt) * 16 + lh;
#pragma unroll
        for (int kc = 0; kc < 8; ++kc) {
            const __bf16* p = wih_bf + col * IN_DIM + kc * 32 + half * 8;
            Bf[nt][kc] = cat8(*(const bf16x8*)p, *(const bf16x8*)(p + 16));
        }
    }

    // Accumulators initialized with folded bias (Wih_b + Whh_b), col-dependent.
    f32x8 acc[2][2];
#pragma unroll
    for (int nt = 0; nt < 2; ++nt) {
        const int col = (n0 + nt) * 16 + lh;
        const float b = bih[col] + bhh[col];
#pragma unroll
        for (int mt = 0; mt < 2; ++mt)
#pragma unroll
            for (int v = 0; v < 8; ++v) acc[mt][nt][v] = b;
    }

    const float* xt = x + (size_t)t * (MB * IN_DIM);
#pragma unroll
    for (int kc = 0; kc < 8; ++kc) {
        bf16x16 Af[2];
#pragma unroll
        for (int mt = 0; mt < 2; ++mt) {
            const int row = (m0 + mt) * 16 + lh;
            const float* p = xt + row * IN_DIM + kc * 32 + half * 8;
            f32x4 a0 = *(const f32x4*)(p);
            f32x4 a1 = *(const f32x4*)(p + 4);
            f32x4 a2 = *(const f32x4*)(p + 16);
            f32x4 a3 = *(const f32x4*)(p + 20);
            bf16x8 lo, hi;
#pragma unroll
            for (int j = 0; j < 4; ++j) {
                lo[j]     = (__bf16)a0[j];
                lo[4 + j] = (__bf16)a1[j];
                hi[j]     = (__bf16)a2[j];
                hi[4 + j] = (__bf16)a3[j];
            }
            Af[mt] = cat8(lo, hi);
        }
#pragma unroll
        for (int mt = 0; mt < 2; ++mt)
#pragma unroll
            for (int nt = 0; nt < 2; ++nt)
                acc[mt][nt] = __builtin_amdgcn_wmma_f32_16x16x32_bf16(
                    false, Af[mt], false, Bf[nt][kc], (short)0, acc[mt][nt],
                    false, false);
    }

    float* ot = out + (size_t)t * (MB * H_DIM);
#pragma unroll
    for (int mt = 0; mt < 2; ++mt)
#pragma unroll
        for (int nt = 0; nt < 2; ++nt) {
            const int col = (n0 + nt) * 16 + lh;
#pragma unroll
            for (int v = 0; v < 8; ++v) {
                const int row = (m0 + mt) * 16 + half * 8 + v;
                ot[row * H_DIM + col] = acc[mt][nt][v];
            }
        }
}

// ---------------------------------------------------------------------------
// Kernel 2 (serial recurrence): single 512-thread workgroup, 2048 steps.
// h kept double-buffered in LDS (bf16); Whh fragments persistent in VGPRs.
// h_t = tanh(a_t + h_{t-1} @ Whh^T); a_t read from (and h_t written back to)
// the same d_out slab. Last step also writes the h_last tail.
// ---------------------------------------------------------------------------
__global__ __launch_bounds__(512, 4) void rnn_phase2_kernel(
        const unsigned short* __restrict__ whh_u16,
        float* __restrict__ out) {
    const __bf16* __restrict__ whh_bf = reinterpret_cast<const __bf16*>(whh_u16);

    __shared__ __align__(16) __bf16 hbuf[2][MB][H_DIM];   // 64 KB

    const int wave = threadIdx.x >> 5;
    const int lane = threadIdx.x & 31;
    const int lh   = lane & 15;
    const int half = lane >> 4;
    const int m0   = (wave >> 3) * 2;
    const int n0   = (wave & 7) * 2;

    // h0 = 0
    for (int i = threadIdx.x; i < MB * H_DIM; i += 512)
        hbuf[0][i >> 8][i & 255] = (__bf16)0.0f;

    // Preload Whh B fragments (B[k,n] = Whh[n,k]: table order per lane).
    bf16x16 Bf[2][8];
#pragma unroll
    for (int nt = 0; nt < 2; ++nt) {
        const int col = (n0 + nt) * 16 + lh;
#pragma unroll
        for (int kc = 0; kc < 8; ++kc) {
            const __bf16* p = whh_bf + col * H_DIM + kc * 32 + half * 8;
            Bf[nt][kc] = cat8(*(const bf16x8*)p, *(const bf16x8*)(p + 16));
        }
    }

    __syncthreads();

#pragma unroll 1
    for (int t = 0; t < T_STEPS; ++t) {
        float* ot = out + (size_t)t * (MB * H_DIM);
        const int rb = t & 1;        // read buffer (h_{t-1})
        const int wb = rb ^ 1;       // write buffer (h_t)

        // acc = a_t (C-matrix layout: 8 rows, one column, per lane)
        f32x8 acc[2][2];
#pragma unroll
        for (int mt = 0; mt < 2; ++mt)
#pragma unroll
            for (int nt = 0; nt < 2; ++nt) {
                const int col = (n0 + nt) * 16 + lh;
#pragma unroll
                for (int v = 0; v < 8; ++v) {
                    const int row = (m0 + mt) * 16 + half * 8 + v;
                    acc[mt][nt][v] = ot[row * H_DIM + col];
                }
            }

        // acc += h_{t-1} @ Whh^T  (A streamed from LDS, B in registers)
#pragma unroll
        for (int kc = 0; kc < 8; ++kc) {
            bf16x16 Af[2];
#pragma unroll
            for (int mt = 0; mt < 2; ++mt) {
                const int row = (m0 + mt) * 16 + lh;
                const __bf16* p = &hbuf[rb][row][kc * 32 + half * 8];
                Af[mt] = cat8(*(const bf16x8*)p, *(const bf16x8*)(p + 16));
            }
#pragma unroll
            for (int mt = 0; mt < 2; ++mt)
#pragma unroll
                for (int nt = 0; nt < 2; ++nt)
                    acc[mt][nt] = __builtin_amdgcn_wmma_f32_16x16x32_bf16(
                        false, Af[mt], false, Bf[nt][kc], (short)0,
                        acc[mt][nt], false, false);
        }

        // h_t = tanh(acc): res[t] (f32, global) + next-step A operand (bf16, LDS)
#pragma unroll
        for (int mt = 0; mt < 2; ++mt)
#pragma unroll
            for (int nt = 0; nt < 2; ++nt) {
                const int col = (n0 + nt) * 16 + lh;
#pragma unroll
                for (int v = 0; v < 8; ++v) {
                    const int row = (m0 + mt) * 16 + half * 8 + v;
                    const float hv = fast_tanh(acc[mt][nt][v]);
                    ot[row * H_DIM + col] = hv;
                    hbuf[wb][row][col] = (__bf16)hv;
                    if (t == T_STEPS - 1)
                        out[(size_t)T_STEPS * MB * H_DIM + row * H_DIM + col] = hv;
                }
            }

        __syncthreads();   // h_t visible in LDS before step t+1 reads it
    }
}

// ---------------------------------------------------------------------------
extern "C" void kernel_launch(void* const* d_in, const int* in_sizes, int n_in,
                              void* d_out, int out_size, void* d_ws, size_t ws_size,
                              hipStream_t stream) {
    const float* x     = (const float*)d_in[0];   // [T, MB, IN]
    const float* wih_w = (const float*)d_in[1];   // [H, IN]
    const float* wih_b = (const float*)d_in[2];   // [H]
    const float* whh_w = (const float*)d_in[3];   // [H, H]
    const float* whh_b = (const float*)d_in[4];   // [H]
    float* out = (float*)d_out;                   // [T*MB*H + MB*H]

    unsigned short* wbf = (unsigned short*)d_ws;  // bf16 Wih | bf16 Whh (256 KB)

    cvt_weights_kernel<<<(2 * H_DIM * IN_DIM + 255) / 256, 256, 0, stream>>>(
        wih_w, whh_w, wbf);

    rnn_phase1_kernel<<<T_STEPS, 512, 0, stream>>>(x, wbf, wih_b, whh_b, out);

    rnn_phase2_kernel<<<1, 512, 0, stream>>>(wbf + (size_t)H_DIM * IN_DIM, out);
}
